// FGAIM_12180527251935
// MI455X (gfx1250) — matrix operand order repaired and checked by reading, not compiled, not run on hardware
//
#include <hip/hip_runtime.h>
#include <math.h>

// ---------------------------------------------------------------------------
// CDNA5 (gfx1250) implementation. All heavy matmuls go through a bf16 WMMA
// GEMM (v_wmma_f32_16x16x32_bf16, f32 accumulate). Graph ops (mean agg,
// segment max/softmax, scatters) are scalar grid-stride kernels.
//
// GEMM tiling: block = 128 rows x 64 cols, K-step 32. 8 waves arranged 4x2;
// each wave computes a 32x32 output tile = 4 WMMA tiles from 2 A fragments
// and 2 B fragments (~10.7 MACs per global byte, 4 WMMAs per 32 LDS dword reads).
// ---------------------------------------------------------------------------

typedef __attribute__((ext_vector_type(16))) __bf16 v16bf;
typedef __attribute__((ext_vector_type(8)))  float  v8f;

union FragAB { v16bf v; unsigned int u[8]; };
union FragC  { v8f   v; float        f[8]; };

__device__ __forceinline__ unsigned short f2bf(float f) {
  unsigned int u = __float_as_uint(f);
  u += 0x7fffu + ((u >> 16) & 1u);        // round-to-nearest-even
  return (unsigned short)(u >> 16);
}
__device__ __forceinline__ unsigned fenc(float f) {   // monotonic float->uint
  unsigned u = __float_as_uint(f);
  return (u & 0x80000000u) ? ~u : (u | 0x80000000u);
}
__device__ __forceinline__ float fdec(unsigned u) {
  unsigned v = (u & 0x80000000u) ? (u & 0x7fffffffu) : ~u;
  return __uint_as_float(v);
}
__device__ __forceinline__ float sigmoidf_(float x) { return 1.f / (1.f + expf(-x)); }

#define LDST 34   // padded LDS row stride (in bf16 elems) for a 32-wide K tile

// C[N x M] = act( A[N x K] @ W[M x K]^T  (+bias) (+addend) (+C if acc) )
// Requires: M % 64 == 0, K % 32 == 0. Rows beyond N are guarded.
__global__ __launch_bounds__(256) void k_wmma_gemm(
    const float* __restrict__ A, const float* __restrict__ W,
    const float* __restrict__ bias, const float* __restrict__ addend,
    float* __restrict__ C, int N, int K, int M, int act, int acc)
{
  __shared__ unsigned short lds_a[128 * LDST];
  __shared__ unsigned short lds_w[64 * LDST];

  const int tid     = threadIdx.x;
  const int rowBase = blockIdx.y * 128;
  const int colBase = blockIdx.x * 64;

  const int wv   = tid >> 5;       // wave 0..7
  const int wr   = wv & 3;         // 4 row groups of 32
  const int wc   = wv >> 2;        // 2 col groups of 32
  const int lane = tid & 31;
  const int hl   = lane >> 4;      // lane half
  const int l16  = lane & 15;

  FragC acc00, acc01, acc10, acc11;  // [row tile][col tile]
#pragma unroll
  for (int i = 0; i < 8; ++i) {
    acc00.f[i] = 0.f; acc01.f[i] = 0.f; acc10.f[i] = 0.f; acc11.f[i] = 0.f;
  }

  // cooperative loader indices
  const int ldrA = tid >> 1;          // 0..127
  const int ldcA = (tid & 1) << 4;    // 0, 16
  const int ldrW = tid >> 2;          // 0..63
  const int ldcW = (tid & 3) << 3;    // 0,8,16,24

  for (int kt = 0; kt < K; kt += 32) {
    // ---- cooperative tile load with f32 -> bf16 conversion ----
    {
      int gr = rowBase + ldrA;
      unsigned short* da = &lds_a[ldrA * LDST + ldcA];
      if (gr < N) {
        const float* s = A + (size_t)gr * K + kt + ldcA;
#pragma unroll
        for (int i = 0; i < 16; ++i) da[i] = f2bf(s[i]);
        if (kt + 32 < K) __builtin_prefetch(s + 32, 0, 1);   // global_prefetch_b8
      } else {
#pragma unroll
        for (int i = 0; i < 16; ++i) da[i] = 0;
      }
      int gw = colBase + ldrW;                       // always < M (M%64==0)
      const float* sw = W + (size_t)gw * K + kt + ldcW;
      unsigned short* dw = &lds_w[ldrW * LDST + ldcW];
#pragma unroll
      for (int i = 0; i < 8; ++i) dw[i] = f2bf(sw[i]);
      if (kt + 32 < K) __builtin_prefetch(sw + 32, 0, 1);
    }
    __syncthreads();

    // ---- fragment assembly per ISA 7.12.2 bf16 layouts ----
    FragAB fa0, fa1, fb0, fb1;
#pragma unroll
    for (int v = 0; v < 8; ++v) {
      int ka = ((v >> 2) << 4) + (hl << 3) + ((v & 3) << 1);   // A: 16x32
      fa0.u[v] = *(const unsigned int*)&lds_a[(wr * 32 +      l16) * LDST + ka];
      fa1.u[v] = *(const unsigned int*)&lds_a[(wr * 32 + 16 + l16) * LDST + ka];
      int kb = ((v >> 2) << 4) + (hl << 3) + ((v & 3) << 1);   // B = W^T: 32x16
      fb0.u[v] = *(const unsigned int*)&lds_w[(wc * 32 +      l16) * LDST + kb];
      fb1.u[v] = *(const unsigned int*)&lds_w[(wc * 32 + 16 + l16) * LDST + kb];
    }
    acc00.v = __builtin_amdgcn_wmma_f32_16x16x32_bf16(false, fa0.v, false, fb0.v,
                                                      (short)0, acc00.v, false, false);
    acc01.v = __builtin_amdgcn_wmma_f32_16x16x32_bf16(false, fa0.v, false, fb1.v,
                                                      (short)0, acc01.v, false, false);
    acc10.v = __builtin_amdgcn_wmma_f32_16x16x32_bf16(false, fa1.v, false, fb0.v,
                                                      (short)0, acc10.v, false, false);
    acc11.v = __builtin_amdgcn_wmma_f32_16x16x32_bf16(false, fa1.v, false, fb1.v,
                                                      (short)0, acc11.v, false, false);
    __syncthreads();
  }

  // ---- epilogue: D layout VGPR r -> row (hl*8 + r), col l16 ----
#pragma unroll
  for (int ti = 0; ti < 2; ++ti) {
#pragma unroll
    for (int tj = 0; tj < 2; ++tj) {
      FragC* a = ti ? (tj ? &acc11 : &acc10) : (tj ? &acc01 : &acc00);
      int gc = colBase + wc * 32 + tj * 16 + l16;
#pragma unroll
      for (int r = 0; r < 8; ++r) {
        int gr = rowBase + wr * 32 + ti * 16 + hl * 8 + r;
        if (gr < N) {
          size_t off = (size_t)gr * M + gc;
          float val = a->f[r];
          if (bias)   val += bias[gc];
          if (addend) val += addend[off];
          if (acc)    val += C[off];
          if (act == 1)      val = fmaxf(val, 0.f);
          else if (act == 2) val = tanhf(val);
          C[off] = val;
        }
      }
    }
  }
}

// ------------------------- scalar support kernels --------------------------
__global__ void k_zero(float* __restrict__ p, long long n) {
  long long st = (long long)gridDim.x * blockDim.x;
  for (long long i = (long long)blockIdx.x * blockDim.x + threadIdx.x; i < n; i += st) p[i] = 0.f;
}
__global__ void k_fillu(unsigned* __restrict__ p, long long n, unsigned v) {
  long long st = (long long)gridDim.x * blockDim.x;
  for (long long i = (long long)blockIdx.x * blockDim.x + threadIdx.x; i < n; i += st) p[i] = v;
}
__global__ void k_copy(float* __restrict__ d, const float* __restrict__ s, long long n) {
  long long st = (long long)gridDim.x * blockDim.x;
  for (long long i = (long long)blockIdx.x * blockDim.x + threadIdx.x; i < n; i += st) d[i] = s[i];
}
__global__ void k_deg(const int* __restrict__ dst, float* __restrict__ deg, long long E) {
  long long st = (long long)gridDim.x * blockDim.x;
  for (long long e = (long long)blockIdx.x * blockDim.x + threadIdx.x; e < E; e += st)
    atomicAdd(&deg[dst[e]], 1.f);
}
__global__ void k_scatter_add(const float* __restrict__ x, const int* __restrict__ src,
                              const int* __restrict__ dst, float* __restrict__ out,
                              long long E, int K) {
  long long n = E * (long long)K;
  long long st = (long long)gridDim.x * blockDim.x;
  for (long long i = (long long)blockIdx.x * blockDim.x + threadIdx.x; i < n; i += st) {
    long long e = i / K; int k = (int)(i - e * K);
    atomicAdd(&out[(size_t)dst[e] * K + k], x[(size_t)src[e] * K + k]);
  }
}
__global__ void k_divdeg(float* __restrict__ m, const float* __restrict__ deg,
                         long long N, int K) {
  long long n = N * (long long)K;
  long long st = (long long)gridDim.x * blockDim.x;
  for (long long i = (long long)blockIdx.x * blockDim.x + threadIdx.x; i < n; i += st) {
    long long r = i / K;
    m[i] /= fmaxf(deg[r], 1.f);
  }
}
__global__ void k_gather_rows(const float* __restrict__ x, const int* __restrict__ rows,
                              float* __restrict__ out, long long R, int K) {
  long long n = R * (long long)K;
  long long st = (long long)gridDim.x * blockDim.x;
  for (long long i = (long long)blockIdx.x * blockDim.x + threadIdx.x; i < n; i += st) {
    long long r = i / K; int k = (int)(i - r * K);
    out[i] = x[(size_t)rows[r] * K + k];
  }
}
__global__ void k_scatter_rows(float* __restrict__ dstm, const int* __restrict__ idx,
                               const float* __restrict__ srcm, long long R, int K) {
  long long n = R * (long long)K;
  long long st = (long long)gridDim.x * blockDim.x;
  for (long long i = (long long)blockIdx.x * blockDim.x + threadIdx.x; i < n; i += st) {
    long long r = i / K; int k = (int)(i - r * K);
    dstm[(size_t)idx[r] * K + k] = srcm[i];
  }
}
__global__ void k_segmax(const float* __restrict__ x, const int* __restrict__ rows,
                         const int* __restrict__ seg, unsigned* __restrict__ out,
                         long long R, int K) {
  long long n = R * (long long)K;
  long long st = (long long)gridDim.x * blockDim.x;
  for (long long i = (long long)blockIdx.x * blockDim.x + threadIdx.x; i < n; i += st) {
    long long r = i / K; int k = (int)(i - r * K);
    int row = rows ? rows[r] : (int)r;
    atomicMax(&out[(size_t)seg[r] * K + k], fenc(x[(size_t)row * K + k]));
  }
}
__global__ void k_decodeu(const unsigned* __restrict__ u, float* __restrict__ f, long long n) {
  long long st = (long long)gridDim.x * blockDim.x;
  for (long long i = (long long)blockIdx.x * blockDim.x + threadIdx.x; i < n; i += st)
    f[i] = fdec(u[i]);
}
__global__ void k_rowdot(const float* __restrict__ X, const float* __restrict__ w,
                         const float* __restrict__ b, float* __restrict__ out,
                         long long R, int K) {
  long long wtot = (long long)gridDim.x * (blockDim.x >> 5);
  long long wid = (long long)blockIdx.x * (blockDim.x >> 5) + (threadIdx.x >> 5);
  int lane = threadIdx.x & 31;
  for (long long r = wid; r < R; r += wtot) {
    float s = 0.f;
    for (int k = lane; k < K; k += 32) s += X[(size_t)r * K + k] * w[k];
    for (int off = 16; off > 0; off >>= 1) s += __shfl_down(s, off, 32);
    if (lane == 0) out[r] = s + b[0];
  }
}
__global__ void k_segmax1(const float* __restrict__ v, const int* __restrict__ seg,
                          unsigned* __restrict__ out, long long R) {
  long long st = (long long)gridDim.x * blockDim.x;
  for (long long i = (long long)blockIdx.x * blockDim.x + threadIdx.x; i < R; i += st)
    atomicMax(&out[seg[i]], fenc(v[i]));
}
__global__ void k_expsum(const float* __restrict__ lg, const int* __restrict__ seg,
                         const float* __restrict__ m, float* __restrict__ att,
                         float* __restrict__ s, long long R) {
  long long st = (long long)gridDim.x * blockDim.x;
  for (long long i = (long long)blockIdx.x * blockDim.x + threadIdx.x; i < R; i += st) {
    float e = expf(lg[i] - m[seg[i]]);
    att[i] = e;
    atomicAdd(&s[seg[i]], e);
  }
}
__global__ void k_norm(float* __restrict__ att, const int* __restrict__ seg,
                       const float* __restrict__ s, long long R) {
  long long st = (long long)gridDim.x * blockDim.x;
  for (long long i = (long long)blockIdx.x * blockDim.x + threadIdx.x; i < R; i += st)
    att[i] /= s[seg[i]];
}
__global__ void k_scatter1(float* __restrict__ ea, const int* __restrict__ pos,
                           const float* __restrict__ att, const int* __restrict__ res,
                           long long n) {
  long long st = (long long)gridDim.x * blockDim.x;
  for (long long i = (long long)blockIdx.x * blockDim.x + threadIdx.x; i < n; i += st)
    ea[pos[i]] = att[res[i]];
}
__global__ __launch_bounds__(512) void k_lstm(
    const float* __restrict__ x, const float* __restrict__ Wih,
    const float* __restrict__ Whh, const float* __restrict__ bih,
    const float* __restrict__ bhh, float* __restrict__ glo) {
  __shared__ float h[128], c[128], g[512];
  int t = threadIdx.x;
  if (t < 128) { h[t] = 0.f; c[t] = 0.f; }
  __syncthreads();
  for (int step = 0; step < 64; ++step) {
    const float* xt = x + step * 512;
    float a = bih[t] + bhh[t];
    const float* wi = Wih + (size_t)t * 512;
    for (int k = 0; k < 512; ++k) a += wi[k] * xt[k];
    const float* wh = Whh + (size_t)t * 128;
    for (int k = 0; k < 128; ++k) a += wh[k] * h[k];
    g[t] = a;
    __syncthreads();
    if (t < 128) {
      float ig = sigmoidf_(g[t]);
      float fg = sigmoidf_(g[128 + t]);
      float gg = tanhf(g[256 + t]);
      float og = sigmoidf_(g[384 + t]);
      float cn = fg * c[t] + ig * gg;
      float hn = og * tanhf(cn);
      c[t] = cn; h[t] = hn;
      glo[step * 128 + t] = fmaxf(hn, 0.f);   // relu(hs)
    }
    __syncthreads();
  }
}
__global__ void k_max3concat(const float* __restrict__ h2, const int* __restrict__ dnode,
                             const float* __restrict__ xcd, const float* __restrict__ glo,
                             const float* __restrict__ x2t, float* __restrict__ xcin) {
  int i = blockIdx.x * blockDim.x + threadIdx.x;
  if (i >= 64 * 256) return;
  int b = i >> 8, j = i & 255;
  float v;
  if (j < 128) {
    float a = h2[(size_t)dnode[b] * 128 + j];
    v = fmaxf(fmaxf(a, xcd[b * 128 + j]), glo[b * 128 + j]);
  } else {
    v = x2t[b * 128 + (j - 128)];
  }
  xcin[i] = v;
}
__global__ void k_outlayer(const float* __restrict__ xc, const float* __restrict__ Wo,
                           const float* __restrict__ bo, float* __restrict__ out) {
  int t = threadIdx.x;
  if (t >= 128) return;
  int b = t >> 1, o = t & 1;
  float s = bo[o];
  const float* xr = xc + (size_t)b * 512;
  const float* wr = Wo + (size_t)o * 512;
  for (int k = 0; k < 512; ++k) s += xr[k] * wr[k];
  out[b * 2 + o] = s;
}

// ------------------------------- host side ---------------------------------
extern "C" void kernel_launch(void* const* d_in, const int* in_sizes, int n_in,
                              void* d_out, int out_size, void* d_ws, size_t ws_size,
                              hipStream_t stream)
{
  (void)out_size; (void)ws_size;
  if (n_in < 56) return;

  // input mapping: dict insertion order, params dict expanded in-place
  const float* result   = (const float*)d_in[0];
  const float* x2       = (const float*)d_in[1];
  const float* glo_fea  = (const float*)d_in[2];
  const float* edge_attr= (const float*)d_in[3];
  const float* c1d_Wl = (const float*)d_in[4];  const float* c1d_bl = (const float*)d_in[5];
  const float* c1d_Wr = (const float*)d_in[6];
  const float* c2d_Wl = (const float*)d_in[7];  const float* c2d_bl = (const float*)d_in[8];
  const float* c2d_Wr = (const float*)d_in[9];
  const float* rbd_Wl = (const float*)d_in[10]; const float* rbd_bl = (const float*)d_in[11];
  const float* rbd_Wr = (const float*)d_in[12]; const float* rbd_Wln= (const float*)d_in[13];
  const float* fg1d_W = (const float*)d_in[14]; const float* fg1d_b = (const float*)d_in[15];
  const float* fg2d_W = (const float*)d_in[16]; const float* fg2d_b = (const float*)d_in[17];
  const float* fg3d_W = (const float*)d_in[18]; const float* fg3d_b = (const float*)d_in[19];
  const float* fl_W   = (const float*)d_in[20]; const float* fl_b   = (const float*)d_in[21];
  const float* sl_W   = (const float*)d_in[22]; const float* sl_b   = (const float*)d_in[23];
  const float* c1t_Wl = (const float*)d_in[24]; const float* c1t_bl = (const float*)d_in[25];
  const float* c1t_Wr = (const float*)d_in[26];
  const float* c2t_Wl = (const float*)d_in[27]; const float* c2t_bl = (const float*)d_in[28];
  const float* c2t_Wr = (const float*)d_in[29];
  const float* rbt_Wl = (const float*)d_in[30]; const float* rbt_bl = (const float*)d_in[31];
  const float* rbt_Wr = (const float*)d_in[32]; const float* rbt_Wln= (const float*)d_in[33];
  const float* fg1t_W = (const float*)d_in[34]; const float* fg1t_b = (const float*)d_in[35];
  const float* fg2t_W = (const float*)d_in[36]; const float* fg2t_b = (const float*)d_in[37];
  const float* fc1_W  = (const float*)d_in[38]; const float* fc1_b  = (const float*)d_in[39];
  const float* fc2_W  = (const float*)d_in[40]; const float* fc2_b  = (const float*)d_in[41];
  const float* out_W  = (const float*)d_in[42]; const float* out_b  = (const float*)d_in[43];
  const float* lstm_Wih = (const float*)d_in[44];
  const float* lstm_Whh = (const float*)d_in[45];
  const float* lstm_bih = (const float*)d_in[46];
  const float* lstm_bhh = (const float*)d_in[47];
  const int* dei     = (const int*)d_in[48];
  const int* dbatch  = (const int*)d_in[49];
  const int* pei     = (const int*)d_in[50];
  const int* res_seg = (const int*)d_in[51];
  const int* dnode   = (const int*)d_in[52];
  const int* rnode   = (const int*)d_in[53];
  const int* depos   = (const int*)d_in[54];
  const int* deres   = (const int*)d_in[55];

  const long long Nd   = in_sizes[0] / 512;
  const long long Nt   = in_sizes[1] / 512;
  const long long Et   = in_sizes[3];
  const long long Ed   = in_sizes[48] / 2;
  const long long Ep   = in_sizes[50] / 2;
  const long long Nres = in_sizes[51];
  const int* dsrc = dei;         const int* ddst = dei + Ed;
  const int* psrc = pei;         const int* pdst = pei + Ep;

  // workspace bump allocator
  char* wp = (char*)d_ws;
  auto alloc = [&](size_t bytes) -> void* {
    void* r = (void*)wp; wp += (bytes + 255) & ~(size_t)255; return r;
  };
  long long BIGF = Nt * 512; if (Nd * 1024 > BIGF) BIGF = Nd * 1024;
  if (Nres * 512 > BIGF) BIGF = Nres * 512;
  float*    R0   = (float*)alloc((size_t)BIGF * 4);
  float*    R1   = (float*)alloc((size_t)BIGF * 4);
  float*    R2   = (float*)alloc((size_t)BIGF * 4);
  float*    degd = (float*)alloc((size_t)Nd * 4);
  float*    degt = (float*)alloc((size_t)Nt * 4);
  unsigned* pu   = (unsigned*)alloc(64 * 1024 * 4);
  float* pooled  = (float*)alloc(64 * 1024 * 4);
  float* xg1     = (float*)alloc(64 * 1024 * 4);
  float* xg      = (float*)alloc(64 * 512 * 4);
  float* xcd     = (float*)alloc(64 * 128 * 4);
  float* glo     = (float*)alloc(64 * 128 * 4);
  float* logits  = (float*)alloc((size_t)Nres * 4);
  float* att     = (float*)alloc((size_t)Nres * 4);
  unsigned* smu  = (unsigned*)alloc(64 * 4);
  float* sm      = (float*)alloc(64 * 4);
  float* ss      = (float*)alloc(64 * 4);
  float* pooledt = (float*)alloc(64 * 128 * 4);
  float* x2t1    = (float*)alloc(64 * 1024 * 4);
  float* x2t     = (float*)alloc(64 * 128 * 4);
  float* xcin    = (float*)alloc(64 * 256 * 4);
  float* xc1     = (float*)alloc(64 * 1024 * 4);

  auto blk = [](long long n) -> int {
    long long b = (n + 255) / 256; if (b < 1) b = 1; if (b > 16384) b = 16384; return (int)b;
  };
  auto gemm = [&](const float* A, const float* W, const float* bias, const float* addend,
                  float* C, long long N, int K, int M, int act, int acc) {
    dim3 g((unsigned)(M / 64), (unsigned)((N + 127) / 128));
    k_wmma_gemm<<<g, 256, 0, stream>>>(A, W, bias, addend, C, (int)N, K, M, act, acc);
  };
  // SAGEConv: out = act( mean @ Wl^T + bl + x @ Wr^T )
  auto sage = [&](const float* xin, int K, int Mo, const float* Wl, const float* bl,
                  const float* Wr, long long Nn, const int* src, const int* dst,
                  long long E, const float* deg, float* mean, float* outb, int act) {
    k_zero<<<blk(Nn * K), 256, 0, stream>>>(mean, Nn * (long long)K);
    k_scatter_add<<<blk(E * K), 256, 0, stream>>>(xin, src, dst, mean, E, K);
    k_divdeg<<<blk(Nn * K), 256, 0, stream>>>(mean, deg, Nn, K);
    gemm(mean, Wl, bl, nullptr, outb, Nn, K, Mo, 0, 0);
    gemm(xin,  Wr, nullptr, nullptr, outb, Nn, K, Mo, act, 1);
  };
  // rblock: x = relu( relu(sage(x)) @ Wln^T + x )
  auto rblock = [&](float* xbuf, float* tmp, float* mean, int D,
                    const float* Wl, const float* bl, const float* Wr, const float* Wln,
                    long long Nn, const int* src, const int* dst, long long E,
                    const float* deg) {
    sage(xbuf, D, D, Wl, bl, Wr, Nn, src, dst, E, deg, mean, tmp, 1);
    gemm(tmp, Wln, nullptr, xbuf, xbuf, Nn, D, D, 1, 0);
  };

  // ---------------- drug branch ----------------
  k_zero<<<blk(Nd), 256, 0, stream>>>(degd, Nd);
  k_deg<<<blk(Ed), 256, 0, stream>>>(ddst, degd, Ed);
  sage(result, 512, 512, c1d_Wl, c1d_bl, c1d_Wr, Nd, dsrc, ddst, Ed, degd, R2, R1, 1);
  sage(R1, 512, 1024, c2d_Wl, c2d_bl, c2d_Wr, Nd, dsrc, ddst, Ed, degd, R2, R0, 1);
  for (int i = 0; i < 4; ++i)
    rblock(R0, R1, R2, 1024, rbd_Wl, rbd_bl, rbd_Wr, rbd_Wln, Nd, dsrc, ddst, Ed, degd);
  k_fillu<<<blk(64 * 1024), 256, 0, stream>>>(pu, 64 * 1024, 0u);
  k_segmax<<<blk(Nd * 1024), 256, 0, stream>>>(R0, nullptr, dbatch, pu, Nd, 1024);
  k_decodeu<<<blk(64 * 1024), 256, 0, stream>>>(pu, pooled, 64 * 1024);
  gemm(pooled, fg1d_W, fg1d_b, nullptr, xg1, 64, 1024, 1024, 1, 0);
  gemm(xg1, fg2d_W, fg2d_b, nullptr, xg, 64, 1024, 512, 0, 0);
  gemm(xg, fg3d_W, fg3d_b, nullptr, xcd, 64, 512, 128, 1, 0);

  // ---------------- LSTM over glo_fea ----------------
  k_lstm<<<1, 512, 0, stream>>>(glo_fea, lstm_Wih, lstm_Whh, lstm_bih, lstm_bhh, glo);

  // ---------------- residue attention -> edge_attr_new ----------------
  k_gather_rows<<<blk(Nres * 512), 256, 0, stream>>>(x2, rnode, R0, Nres, 512);
  gemm(R0, fl_W, fl_b, nullptr, R1, Nres, 512, 512, 2, 0);     // tanh
  {
    long long rb = (Nres + 7) / 8; if (rb > 16384) rb = 16384; if (rb < 1) rb = 1;
    k_rowdot<<<(int)rb, 256, 0, stream>>>(R1, sl_W, sl_b, logits, Nres, 512);
  }
  k_fillu<<<1, 64, 0, stream>>>(smu, 64, 0u);
  k_segmax1<<<blk(Nres), 256, 0, stream>>>(logits, res_seg, smu, Nres);
  k_decodeu<<<1, 64, 0, stream>>>(smu, sm, 64);
  k_zero<<<1, 64, 0, stream>>>(ss, 64);
  k_expsum<<<blk(Nres), 256, 0, stream>>>(logits, res_seg, sm, att, ss, Nres);
  k_norm<<<blk(Nres), 256, 0, stream>>>(att, res_seg, ss, Nres);
  float* eout = (float*)d_out + 128 + 64 * 512;
  k_copy<<<blk(Et), 256, 0, stream>>>(eout, edge_attr, Et);
  k_scatter1<<<blk(2 * Nres), 256, 0, stream>>>(eout, depos, att, deres, 2 * Nres);

  // ---------------- protein branch ----------------
  k_copy<<<blk(Nt * 512), 256, 0, stream>>>(R0, x2, Nt * 512);          // x2u
  k_scatter_rows<<<blk(64 * 512), 256, 0, stream>>>(R0, dnode, xg, 64, 512);
  k_zero<<<blk(Nt), 256, 0, stream>>>(degt, Nt);
  k_deg<<<blk(Ep), 256, 0, stream>>>(pdst, degt, Ep);
  sage(R0, 512, 64, c1t_Wl, c1t_bl, c1t_Wr, Nt, psrc, pdst, Ep, degt, R2, R1, 1);
  sage(R1, 64, 128, c2t_Wl, c2t_bl, c2t_Wr, Nt, psrc, pdst, Ep, degt, R2, R0, 1);
  for (int i = 0; i < 4; ++i)
    rblock(R0, R1, R2, 128, rbt_Wl, rbt_bl, rbt_Wr, rbt_Wln, Nt, psrc, pdst, Ep, degt);
  k_fillu<<<blk(64 * 128), 256, 0, stream>>>(pu, 64 * 128, 0u);
  k_segmax<<<blk(Nres * 128), 256, 0, stream>>>(R0, rnode, res_seg, pu, Nres, 128);
  k_decodeu<<<blk(64 * 128), 256, 0, stream>>>(pu, pooledt, 64 * 128);
  gemm(pooledt, fg1t_W, fg1t_b, nullptr, x2t1, 64, 128, 1024, 1, 0);
  gemm(x2t1, fg2t_W, fg2t_b, nullptr, x2t, 64, 1024, 128, 0, 0);

  // ---------------- head ----------------
  k_max3concat<<<blk(64 * 256), 256, 0, stream>>>(R0, dnode, xcd, glo, x2t, xcin);
  gemm(xcin, fc1_W, fc1_b, nullptr, xc1, 64, 256, 1024, 1, 0);
  float* xc_out = (float*)d_out + 128;
  gemm(xc1, fc2_W, fc2_b, nullptr, xc_out, 64, 1024, 512, 1, 0);
  k_outlayer<<<1, 128, 0, stream>>>(xc_out, out_W, out_b, (float*)d_out);
}
